// LSTMModel_10995116277990
// MI455X (gfx1250) — matrix operand order repaired and checked by reading, not compiled, not run on hardware
//
#include <hip/hip_runtime.h>
#include <hip/hip_bf16.h>

typedef __attribute__((ext_vector_type(16))) __bf16 v16bf;
typedef __attribute__((ext_vector_type(8)))  __bf16 v8bf;
typedef __attribute__((ext_vector_type(8)))  float  v8f;

#define ISZ 256
#define HSZ 1024
#define BSZ 256
#define TSZ 512
#define KSZ 1280   /* ISZ + HSZ */

union ABFrag { v16bf v; v8bf h[2]; };

__device__ __forceinline__ float sigmoidf_(float x) { return 1.0f / (1.0f + __expf(-x)); }

// ---- Prologue: pack [W_ih | W_hh] -> bf16, row-major [4096][1280] -----------
__global__ void pack_weights(const float* __restrict__ Wih,
                             const float* __restrict__ Whh,
                             __bf16* __restrict__ Wc) {
    int idx = blockIdx.x * blockDim.x + threadIdx.x;   // 0 .. 4096*1280-1
    int g = idx / KSZ;
    int k = idx % KSZ;
    float v = (k < ISZ) ? Wih[g * ISZ + k] : Whh[g * HSZ + (k - ISZ)];
    Wc[idx] = (__bf16)v;
}

__global__ void add_bias(const float* __restrict__ bih,
                         const float* __restrict__ bhh,
                         float* __restrict__ bias) {
    int i = blockIdx.x * blockDim.x + threadIdx.x;     // 0 .. 4095
    bias[i] = bih[i] + bhh[i];
}

// ---- Prologue: x fp32 -> bf16 (4 elements/thread) ---------------------------
struct bf4 { __bf16 a, b, c, d; };
__global__ void convert_x(const float4* __restrict__ x, bf4* __restrict__ xb) {
    int idx = blockIdx.x * blockDim.x + threadIdx.x;
    float4 v = x[idx];
    bf4 o; o.a = (__bf16)v.x; o.b = (__bf16)v.y; o.c = (__bf16)v.z; o.d = (__bf16)v.w;
    xb[idx] = o;
}

// ---- Prologue: zero h (both buffers) and c ---------------------------------
__global__ void init_state(__bf16* __restrict__ h0, __bf16* __restrict__ h1,
                           float* __restrict__ c) {
    int i = blockIdx.x * blockDim.x + threadIdx.x;     // 0 .. B*H-1
    h0[i] = (__bf16)0.0f;
    h1[i] = (__bf16)0.0f;
    c[i]  = 0.0f;
}

// ---- One LSTM time step: fused GEMM (K=1280, bf16 WMMA) + cell update ------
// Block: 256 threads = 8 waves = {2 row-subtiles x 4 gates}.
// Each wave computes ONE gate's 16x16 tile over full K (two acc chains),
// parks it in LDS; after the barrier the block does the fused cell update
// for its 32 batch rows x 16 hidden cols.
// Grid: (64, 8) -> 1024 hidden cols x 256 batch rows = 4096 waves/step.
__global__ void __launch_bounds__(256)
lstm_step(const __bf16* __restrict__ xb,   // [B][T][I] bf16
          const __bf16* __restrict__ Wc,   // [4H][K] bf16
          const float*  __restrict__ bias, // [4H]
          const __bf16* __restrict__ hin,  // [B][H] bf16 (h_{t-1})
          __bf16*       __restrict__ hout, // [B][H] bf16 (h_t)
          float*        __restrict__ c,    // [B][H] fp32 (in/out)
          int t)
{
    __shared__ float sgate[2][4][256];     // [subrow][gate][m*16+n], 8 KB

    const int lane = threadIdx.x & 31;
    const int wave = threadIdx.x >> 5;
    const int half = lane >> 4;            // lane half selects K sub-range (ISA layout)
    const int l16  = lane & 15;
    const int subrow = wave >> 2;          // 0..1
    const int gate   = wave & 3;           // i, f, g, o
    const int rowbase = blockIdx.y * 32 + subrow * 16;   // batch rows
    const int col     = blockIdx.x * 16 + l16;           // hidden unit j (lane's column)

    const v8f vzero = {0.f, 0.f, 0.f, 0.f, 0.f, 0.f, 0.f, 0.f};
    v8f acc0 = vzero, acc1 = vzero;        // two independent WMMA chains

    // B = W^T: lane's column n of the B tile is W row (gate*H + col);
// lanes 0-15 hold K k0..k0+7,k0+16..k0+23; lanes 16-31 hold K k0+8..k0+15,k0+24..k0+31.
    const __bf16* brow = Wc + (size_t)(gate * HSZ + col) * KSZ + half * 8;

    // ---- K region 1: x_t contribution (K = 0..255) ----
    {
        const __bf16* arow = xb + ((size_t)(rowbase + l16) * TSZ + (size_t)t) * ISZ;
#pragma unroll
        for (int k0 = 0; k0 < ISZ; k0 += 64) {
            ABFrag a0, b0, a1, b1;
            a0.h[0] = *(const v8bf*)(arow + k0 + half * 8);
            a0.h[1] = *(const v8bf*)(arow + k0 + 16 + half * 8);
            b0.h[0] = *(const v8bf*)(brow + k0);
            b0.h[1] = *(const v8bf*)(brow + k0 + 16);
            a1.h[0] = *(const v8bf*)(arow + k0 + 32 + half * 8);
            a1.h[1] = *(const v8bf*)(arow + k0 + 48 + half * 8);
            b1.h[0] = *(const v8bf*)(brow + k0 + 32);
            b1.h[1] = *(const v8bf*)(brow + k0 + 48);
            acc0 = __builtin_amdgcn_wmma_f32_16x16x32_bf16(
                false, a0.v, false, b0.v, (short)0, acc0, false, false);
            acc1 = __builtin_amdgcn_wmma_f32_16x16x32_bf16(
                false, a1.v, false, b1.v, (short)0, acc1, false, false);
        }
    }
    // ---- K region 2: h_{t-1} contribution (K = 256..1279) ----
    {
        const __bf16* arow = hin + (size_t)(rowbase + l16) * HSZ;
#pragma unroll 4
        for (int k0 = 0; k0 < HSZ; k0 += 64) {
            ABFrag a0, b0, a1, b1;
            a0.h[0] = *(const v8bf*)(arow + k0 + half * 8);
            a0.h[1] = *(const v8bf*)(arow + k0 + 16 + half * 8);
            b0.h[0] = *(const v8bf*)(brow + ISZ + k0);
            b0.h[1] = *(const v8bf*)(brow + ISZ + k0 + 16);
            a1.h[0] = *(const v8bf*)(arow + k0 + 32 + half * 8);
            a1.h[1] = *(const v8bf*)(arow + k0 + 48 + half * 8);
            b1.h[0] = *(const v8bf*)(brow + ISZ + k0 + 32);
            b1.h[1] = *(const v8bf*)(brow + ISZ + k0 + 48);
            acc0 = __builtin_amdgcn_wmma_f32_16x16x32_bf16(
                false, a0.v, false, b0.v, (short)0, acc0, false, false);
            acc1 = __builtin_amdgcn_wmma_f32_16x16x32_bf16(
                false, a1.v, false, b1.v, (short)0, acc1, false, false);
        }
    }

    // Park this wave's gate tile in LDS.
    // C/D layout: VGPR r, lanes 0-15 -> M=r, lanes 16-31 -> M=8+r; N = l16.
    {
        float* slab = &sgate[subrow][gate][0];
#pragma unroll
        for (int r = 0; r < 8; ++r)
            slab[(r + 8 * half) * 16 + l16] = acc0[r] + acc1[r];
    }
    __syncthreads();

    // ---- Fused LSTM cell update: 512 cells, 2 per thread ----
#pragma unroll
    for (int it = 0; it < 2; ++it) {
        const int cell = (int)threadIdx.x + it * 256;   // 0..511
        const int sr   = cell >> 8;                     // subrow
        const int idx  = cell & 255;                    // m*16+n
        const int m    = idx >> 4;
        const int n    = idx & 15;
        const int row  = blockIdx.y * 32 + sr * 16 + m;
        const int colj = blockIdx.x * 16 + n;
        const size_t off = (size_t)row * HSZ + colj;
        float iv = sigmoidf_(sgate[sr][0][idx] + bias[0 * HSZ + colj]);
        float fv = sigmoidf_(sgate[sr][1][idx] + bias[1 * HSZ + colj]);
        float gv = tanhf   (sgate[sr][2][idx] + bias[2 * HSZ + colj]);
        float ov = sigmoidf_(sgate[sr][3][idx] + bias[3 * HSZ + colj]);
        float cn = fv * c[off] + iv * gv;
        c[off] = cn;
        hout[off] = (__bf16)(ov * tanhf(cn));
    }
}

// ---- Final FC + sigmoid: out[b] = sigmoid(h[b,:] . W_fc + b_fc) ------------
__global__ void fc_kernel(const __bf16* __restrict__ hb,
                          const float* __restrict__ Wfc,
                          const float* __restrict__ bfc,
                          float* __restrict__ out) {
    __shared__ float red[256];
    const int b = blockIdx.x;
    const int tid = threadIdx.x;
    float s = 0.0f;
    for (int j = tid; j < HSZ; j += 256)
        s += (float)hb[(size_t)b * HSZ + j] * Wfc[j];
    red[tid] = s;
    __syncthreads();
    for (int stride = 128; stride > 0; stride >>= 1) {
        if (tid < stride) red[tid] += red[tid + stride];
        __syncthreads();
    }
    if (tid == 0) out[b] = sigmoidf_(red[0] + bfc[0]);
}

extern "C" void kernel_launch(void* const* d_in, const int* in_sizes, int n_in,
                              void* d_out, int out_size, void* d_ws, size_t ws_size,
                              hipStream_t stream) {
    const float* x   = (const float*)d_in[0];
    const float* Wih = (const float*)d_in[1];
    const float* Whh = (const float*)d_in[2];
    const float* bih = (const float*)d_in[3];
    const float* bhh = (const float*)d_in[4];
    const float* Wfc = (const float*)d_in[5];
    const float* bfc = (const float*)d_in[6];
    float* out = (float*)d_out;

    // Workspace layout (256B-aligned offsets), total ~76 MB
    char* ws = (char*)d_ws;
    __bf16* Wc   = (__bf16*)(ws + 0);            // 4096*1280*2 = 10,485,760
    float*  bias = (float*)(ws + 10485760);      // 4096*4      =     16,384
    __bf16* xbuf = (__bf16*)(ws + 10502144);     // 256*512*256*2 = 67,108,864
    __bf16* hb0  = (__bf16*)(ws + 77611008);     // 256*1024*2  =    524,288
    __bf16* hb1  = (__bf16*)(ws + 78135296);     // 256*1024*2  =    524,288
    float*  cst  = (float*)(ws + 78659584);      // 256*1024*4  =  1,048,576

    pack_weights<<<(4096 * KSZ) / 256, 256, 0, stream>>>(Wih, Whh, Wc);
    add_bias<<<4096 / 256, 256, 0, stream>>>(bih, bhh, bias);
    convert_x<<<(BSZ * TSZ * ISZ / 4) / 256, 256, 0, stream>>>((const float4*)x, (bf4*)xbuf);
    init_state<<<(BSZ * HSZ) / 256, 256, 0, stream>>>(hb0, hb1, cst);

    __bf16* hin = hb0;   // zero-initialized h0
    __bf16* hout = hb1;
    for (int t = 0; t < TSZ; ++t) {
        lstm_step<<<dim3(64, 8), 256, 0, stream>>>(xbuf, Wc, bias, hin, hout, cst, t);
        __bf16* tmp = hin; hin = hout; hout = tmp;
    }
    // After 512 steps, the final h is in `hin`
    fc_kernel<<<BSZ, 256, 0, stream>>>(hin, Wfc, bfc, out);
}
